// _RNNRegressor_53171695125437
// MI455X (gfx1250) — compile-verified, matrix operand order checked by reading, not run on hardware
//
#include <hip/hip_runtime.h>

// ---------------------------------------------------------------------------
// 2-layer LSTM (H=512) + linear head for MI455X (gfx1250, wave32, WMMA).
// Per timestep, one fused kernel per layer computes
//   gates = [x_t | h_{t-1}] @ [W_ih^T ; W_hh^T]   (bf16 WMMA, f32 accum)
// and applies the LSTM cell update in-block via an LDS-staged epilogue
// (weight columns permuted so each block owns all 4 gates of 64 hidden
// units). Weights (~6.5MB bf16) stay resident in the 192MB L2.
// k-loop: 2-deep ping-pong software pipeline with individually named
// fragment registers (no arrays -> global_load_b128, no rotation copies).
// ---------------------------------------------------------------------------

typedef unsigned short u16;
typedef __attribute__((ext_vector_type(16))) __bf16 v16bf;
typedef __attribute__((ext_vector_type(8)))  float  v8f;
typedef __attribute__((ext_vector_type(8)))  unsigned int v8u;
typedef __attribute__((ext_vector_type(4)))  unsigned int v4u;

#define B_   256
#define T_   256
#define DIN_ 64
#define H_   512
#define G4_  2048   // 4*H

// ---- helpers ---------------------------------------------------------------

static __device__ __forceinline__ u16 f2bf(float f) {
  unsigned u = __float_as_uint(f);
  u += 0x7FFFu + ((u >> 16) & 1u);   // round to nearest even
  return (u16)(u >> 16);
}

static __device__ __forceinline__ float sigf(float x) {
  return 1.0f / (1.0f + __expf(-x));
}
static __device__ __forceinline__ float tanhfast(float x) {
  x = fminf(fmaxf(x, -15.0f), 15.0f);
  float e = __expf(2.0f * x);
  return (e - 1.0f) / (e + 1.0f);
}

// Load a 16x32 bf16 WMMA fragment from a per-lane pointer (already offset by
// row and half*8). Per-lane K pattern (ISA 7.12.2): pairs at
// k0 + half*8 + {0,2,4,6} and + {16,18,20,22} -> two b128 loads per lane.
static __device__ __forceinline__ v16bf ldfrag(const u16* __restrict__ p) {
  v4u lo = *(const v4u*)(p);
  v4u hi = *(const v4u*)(p + 16);
  v8u u;
  u[0] = lo[0]; u[1] = lo[1]; u[2] = lo[2]; u[3] = lo[3];
  u[4] = hi[0]; u[5] = hi[1]; u[6] = hi[2]; u[7] = hi[3];
  return __builtin_bit_cast(v16bf, u);
}

#define WMMA_BF16(A, Bv, C) \
  __builtin_amdgcn_wmma_f32_16x16x32_bf16(false, (A), false, (Bv), (short)0, (C), false, false)

// One GEMM phase over nkb (even, >=2) 32-deep k-blocks; A stream from `ap`,
// B columns from bp0..bp3 (all by value, global provenance preserved).
// 2-deep ping-pong software pipeline, no register rotation copies.
static __device__ __forceinline__ void gemm_phase(
    const u16* __restrict__ ap,
    const u16* __restrict__ bp0, const u16* __restrict__ bp1,
    const u16* __restrict__ bp2, const u16* __restrict__ bp3,
    int nkb, v8f& c0, v8f& c1, v8f& c2, v8f& c3) {
  v16bf A0 = ldfrag(ap);
  v16bf B00 = ldfrag(bp0), B01 = ldfrag(bp1), B02 = ldfrag(bp2), B03 = ldfrag(bp3);

#pragma unroll 1
  for (int kb = 0; kb + 2 < nkb; kb += 2) {
    v16bf A1  = ldfrag(ap + 32);
    v16bf B10 = ldfrag(bp0 + 32), B11 = ldfrag(bp1 + 32);
    v16bf B12 = ldfrag(bp2 + 32), B13 = ldfrag(bp3 + 32);
    c0 = WMMA_BF16(A0, B00, c0);
    c1 = WMMA_BF16(A0, B01, c1);
    c2 = WMMA_BF16(A0, B02, c2);
    c3 = WMMA_BF16(A0, B03, c3);
    ap += 64; bp0 += 64; bp1 += 64; bp2 += 64; bp3 += 64;
    A0  = ldfrag(ap);
    B00 = ldfrag(bp0); B01 = ldfrag(bp1); B02 = ldfrag(bp2); B03 = ldfrag(bp3);
    c0 = WMMA_BF16(A1, B10, c0);
    c1 = WMMA_BF16(A1, B11, c1);
    c2 = WMMA_BF16(A1, B12, c2);
    c3 = WMMA_BF16(A1, B13, c3);
  }
  // tail pair
  {
    v16bf A1  = ldfrag(ap + 32);
    v16bf B10 = ldfrag(bp0 + 32), B11 = ldfrag(bp1 + 32);
    v16bf B12 = ldfrag(bp2 + 32), B13 = ldfrag(bp3 + 32);
    c0 = WMMA_BF16(A0, B00, c0);
    c1 = WMMA_BF16(A0, B01, c1);
    c2 = WMMA_BF16(A0, B02, c2);
    c3 = WMMA_BF16(A0, B03, c3);
    c0 = WMMA_BF16(A1, B10, c0);
    c1 = WMMA_BF16(A1, B11, c1);
    c2 = WMMA_BF16(A1, B12, c2);
    c3 = WMMA_BF16(A1, B13, c3);
  }
}

// ---- pack kernels ----------------------------------------------------------

__global__ void k_pack_x(const float* __restrict__ x, u16* __restrict__ xbf) {
  size_t total = (size_t)B_ * T_ * DIN_;
  for (size_t i = (size_t)blockIdx.x * blockDim.x + threadIdx.x; i < total;
       i += (size_t)gridDim.x * blockDim.x)
    xbf[i] = f2bf(x[i]);
}

// Build Wcat[n'][k], n' permuted so block bx (256 cols) covers gates q=0..3
// of hidden units j in [bx*64, bx*64+64):  n' = bx*256 + q*64 + jl,
// source row g = q*512 + bx*64 + jl.  Cols: [0,Ki) from Wi, [Ki,Ki+Kh) from Wh.
__global__ void k_pack_cat(const float* __restrict__ Wi, int Ki,
                           const float* __restrict__ Wh, int Kh,
                           u16* __restrict__ out) {
  int KT = Ki + Kh;
  size_t total = (size_t)G4_ * KT;
  for (size_t idx = (size_t)blockIdx.x * blockDim.x + threadIdx.x; idx < total;
       idx += (size_t)gridDim.x * blockDim.x) {
    int np = (int)(idx / KT);
    int k  = (int)(idx % KT);
    int jb = np >> 8, r = np & 255, q = r >> 6, jl = r & 63;
    int g  = q * H_ + jb * 64 + jl;
    float v = (k < Ki) ? Wi[(size_t)g * Ki + k] : Wh[(size_t)g * Kh + (k - Ki)];
    out[idx] = f2bf(v);
  }
}

__global__ void k_pack_bias(const float* bi0, const float* bh0,
                            const float* bi1, const float* bh1,
                            float* __restrict__ bias0, float* __restrict__ bias1) {
  int i = blockIdx.x * blockDim.x + threadIdx.x;
  if (i < G4_) bias0[i] = bi0[i] + bh0[i];
  else if (i < 2 * G4_) bias1[i - G4_] = bi1[i - G4_] + bh1[i - G4_];
}

__global__ void k_zero16(uint4* __restrict__ p, size_t n16) {
  uint4 z = {0u, 0u, 0u, 0u};
  for (size_t i = (size_t)blockIdx.x * blockDim.x + threadIdx.x; i < n16;
       i += (size_t)gridDim.x * blockDim.x)
    p[i] = z;
}

// ---- fused LSTM step: GEMM (WMMA) + cell update ----------------------------
// grid (8,8), block 256 (8 waves). blockIdx.y -> 32 batch rows,
// blockIdx.x -> 64 hidden units (all 4 gates). Wave tile 16(M) x 64(N).
__global__ __launch_bounds__(256) void k_lstm_step(
    const u16* __restrict__ src0, int s0stride, int K0,  // x-part rows [B,K0]
    const u16* __restrict__ hprev,                       // [B,H] bf16
    const u16* __restrict__ Wcat,                        // [G4][K0+H] permuted
    const float* __restrict__ bias,                      // [G4] gate-major
    float* __restrict__ cstate,                          // [B,H]
    u16* __restrict__ hnext,                             // [B,H] bf16 out
    float* __restrict__ hf32)                            // optional f32 h out
{
  __shared__ float smem[4][32][72];  // [gate][m][j] pre-activations

  const int tid  = threadIdx.x;
  const int lane = tid & 31;
  const int wave = tid >> 5;
  const int wm   = wave >> 2;              // 0..1
  const int wn   = wave & 3;               // 0..3 == gate q
  const int KTOT = K0 + H_;
  const int nk0  = K0 >> 5;                // k-blocks in x region (even)

  const int m0    = blockIdx.y * 32 + wm * 16;      // batch row base of wave
  const int nrow0 = blockIdx.x * 256 + wn * 64;     // Wcat row base of wave

  // Per-lane fragment base pointers (row + half*8 folded in once).
  const int m = lane & 15, half = lane >> 4;
  const u16* apX = src0  + (size_t)(m0 + m) * (size_t)s0stride + (half << 3);
  const u16* apH = hprev + (size_t)(m0 + m) * (size_t)H_       + (half << 3);
  const u16* bq0 = Wcat + (size_t)(nrow0 +  0 + m) * (size_t)KTOT + (half << 3);
  const u16* bq1 = Wcat + (size_t)(nrow0 + 16 + m) * (size_t)KTOT + (half << 3);
  const u16* bq2 = Wcat + (size_t)(nrow0 + 32 + m) * (size_t)KTOT + (half << 3);
  const u16* bq3 = Wcat + (size_t)(nrow0 + 48 + m) * (size_t)KTOT + (half << 3);

  v8f c0 = {}, c1 = {}, c2 = {}, c3 = {};
  const int xoff = nk0 * 32;
  gemm_phase(apX, bq0, bq1, bq2, bq3, nk0, c0, c1, c2, c3);          // x-part
  gemm_phase(apH, bq0 + xoff, bq1 + xoff, bq2 + xoff, bq3 + xoff,    // h-part
             H_ >> 5, c0, c1, c2, c3);

  // C layout: VGPR v -> M = v + 8*(lane>>4), N = lane&15
  const int ln = lane & 15;
  const int mrow = wm * 16 + 8 * half;
#pragma unroll
  for (int v = 0; v < 8; ++v) smem[wn][mrow + v][ 0 + ln] = c0[v];
#pragma unroll
  for (int v = 0; v < 8; ++v) smem[wn][mrow + v][16 + ln] = c1[v];
#pragma unroll
  for (int v = 0; v < 8; ++v) smem[wn][mrow + v][32 + ln] = c2[v];
#pragma unroll
  for (int v = 0; v < 8; ++v) smem[wn][mrow + v][48 + ln] = c3[v];

  __syncthreads();

  // Cell update: 32 rows x 64 units = 2048 elems, 8 per thread.
#pragma unroll
  for (int rep = 0; rep < 8; ++rep) {
    int idx = rep * 256 + tid;
    int ml = idx >> 6, jl = idx & 63;
    int jg = blockIdx.x * 64 + jl;
    int b  = blockIdx.y * 32 + ml;
    float pi = smem[0][ml][jl] + bias[jg];
    float pf = smem[1][ml][jl] + bias[H_ + jg];
    float pg = smem[2][ml][jl] + bias[2 * H_ + jg];
    float po = smem[3][ml][jl] + bias[3 * H_ + jg];
    float ig = sigf(pi), fg = sigf(pf), gg = tanhfast(pg), og = sigf(po);
    size_t ix = (size_t)b * H_ + jg;
    float c = fg * cstate[ix] + ig * gg;
    cstate[ix] = c;
    float h = og * tanhfast(c);
    hnext[ix] = f2bf(h);
    if (hf32) hf32[ix] = h;
  }
}

// ---- head ------------------------------------------------------------------

__global__ void k_head(const float* __restrict__ h1, const float* __restrict__ w,
                       const float* __restrict__ hb, float* __restrict__ out) {
  int b = blockIdx.x * blockDim.x + threadIdx.x;
  if (b >= B_) return;
  float s = 0.0f;
  for (int j = 0; j < H_; ++j) s = fmaf(h1[(size_t)b * H_ + j], w[j], s);
  out[b] = s + hb[0];
}

// ---- launch ----------------------------------------------------------------

extern "C" void kernel_launch(void* const* d_in, const int* in_sizes, int n_in,
                              void* d_out, int out_size, void* d_ws, size_t ws_size,
                              hipStream_t stream) {
  const float* x    = (const float*)d_in[0];
  const float* Wih0 = (const float*)d_in[1];
  const float* Whh0 = (const float*)d_in[2];
  const float* bih0 = (const float*)d_in[3];
  const float* bhh0 = (const float*)d_in[4];
  const float* Wih1 = (const float*)d_in[5];
  const float* Whh1 = (const float*)d_in[6];
  const float* bih1 = (const float*)d_in[7];
  const float* bhh1 = (const float*)d_in[8];
  const float* hw   = (const float*)d_in[9];
  const float* hb   = (const float*)d_in[10];
  float* out = (float*)d_out;

  // workspace layout (~17.6 MB total)
  char* ws = (char*)d_ws;
  size_t o = 0;
  u16*   xbf   = (u16*)(ws + o);  o += (size_t)B_ * T_ * DIN_ * 2;     // 8.39 MB
  u16*   wcat0 = (u16*)(ws + o);  o += (size_t)G4_ * (DIN_ + H_) * 2;  // 2.36 MB
  u16*   wcat1 = (u16*)(ws + o);  o += (size_t)G4_ * (2 * H_) * 2;     // 4.19 MB
  float* bias0 = (float*)(ws + o); o += G4_ * 4;
  float* bias1 = (float*)(ws + o); o += G4_ * 4;
  char*  zbase = ws + o;
  u16*   h0[2]; u16* h1[2];
  h0[0] = (u16*)(ws + o); o += (size_t)B_ * H_ * 2;
  h0[1] = (u16*)(ws + o); o += (size_t)B_ * H_ * 2;
  h1[0] = (u16*)(ws + o); o += (size_t)B_ * H_ * 2;
  h1[1] = (u16*)(ws + o); o += (size_t)B_ * H_ * 2;
  float* c0  = (float*)(ws + o); o += (size_t)B_ * H_ * 4;
  float* c1  = (float*)(ws + o); o += (size_t)B_ * H_ * 4;
  float* h1f = (float*)(ws + o); o += (size_t)B_ * H_ * 4;
  size_t zbytes = (size_t)(ws + o - zbase);

  // pack / init (re-run every call: deterministic)
  k_pack_x<<<1024, 256, 0, stream>>>(x, xbf);
  k_pack_cat<<<1024, 256, 0, stream>>>(Wih0, DIN_, Whh0, H_, wcat0);
  k_pack_cat<<<1024, 256, 0, stream>>>(Wih1, H_, Whh1, H_, wcat1);
  k_pack_bias<<<(2 * G4_ + 255) / 256, 256, 0, stream>>>(bih0, bhh0, bih1, bhh1,
                                                         bias0, bias1);
  k_zero16<<<512, 256, 0, stream>>>((uint4*)zbase, zbytes / 16);

  dim3 grid(8, 8);
  for (int t = 0; t < T_; ++t) {
    const int pb = t & 1, nb = pb ^ 1;
    // layer 0: A = [x_t | h0]
    k_lstm_step<<<grid, 256, 0, stream>>>(xbf + (size_t)t * DIN_, T_ * DIN_, DIN_,
                                          h0[pb], wcat0, bias0, c0, h0[nb],
                                          (float*)nullptr);
    // layer 1: A = [h0_t | h1]
    k_lstm_step<<<grid, 256, 0, stream>>>(h0[nb], H_, H_,
                                          h1[pb], wcat1, bias1, c1, h1[nb], h1f);
  }
  k_head<<<1, 256, 0, stream>>>(h1f, hw, hb, out);
}
